// PairwisePolicyHead_34050500722847
// MI455X (gfx1250) — compile-verified, matrix operand order checked by reading, not run on hardware
//
#include <hip/hip_runtime.h>
#include <hip/hip_bf16.h>
#include <stdint.h>

typedef __attribute__((ext_vector_type(16))) __bf16 v16bf;
typedef __attribute__((ext_vector_type(8)))  float  v8f;

#define N_NODES 1024
#define DIM     128
#define PAIRS   ((N_NODES * (N_NODES - 1)) / 2)   // 523776 = 64 * 8184 exactly
#define PPB     64                                 // pairs per block (4 m-tiles of 16)
#define XS_STRIDE 132                              // 128 dwords + 4 pad (bank-conflict avoidance)

// Packed f32x2 -> bf16x2 round-to-nearest-even; should lower to v_cvt_pk_bf16_f32.
__device__ __forceinline__ uint32_t pack_bf16x2(float lo, float hi) {
  union { __hip_bfloat162 h; uint32_t u; } cv;
  cv.h = __float22bfloat162_rn(make_float2(lo, hi));
  return cv.u;
}

// ---------------- Kernel 1: A = H @ Wa, B = H @ Wb (f32) ----------------
__global__ void precompute_AB(const float* __restrict__ H,
                              const float* __restrict__ W1,
                              float* __restrict__ A,
                              float* __restrict__ B) {
  const int n = blockIdx.x;
  const int c = threadIdx.x;
  const float* __restrict__ h = H + n * DIM;
  float a = 0.f, b = 0.f;
#pragma unroll 8
  for (int k = 0; k < DIM; ++k) {
    float hv = h[k];
    a = fmaf(hv, W1[k * DIM + c], a);
    b = fmaf(hv, W1[(DIM + k) * DIM + c], b);
  }
  A[n * DIM + c] = a;
  B[n * DIM + c] = b;
}

// ---- Kernel 1b: pack Wcd = W1[256:512] (256x128) into bf16 WMMA fragment layout ----
// Used as the A operand of hidden^T = Wcd^T @ X^T.
// Flat dword f = (((wv*8 + kk)*32) + lane)*8 + r
//   lane 0-15 : M(row)=col = wv*16 + lane,    K = kk*32 + 2r (+1 in hi half of dword)
//   lane16-31 : M(row)=col = wv*16 + lane-16, K = kk*32 + 16 + 2r
__global__ void pack_wfrag(const float* __restrict__ W1,
                           uint32_t* __restrict__ Wfrag) {
  const int f  = blockIdx.x * blockDim.x + threadIdx.x;  // 0..16383
  const int r  = f & 7;
  const int l  = (f >> 3) & 31;
  const int kk = (f >> 8) & 7;
  const int wv = f >> 11;
  const int col = wv * 16 + (l & 15);
  const int K   = kk * 32 + ((l >> 4) * 16) + 2 * r;
  Wfrag[f] = pack_bf16x2(W1[(2 * DIM + K)     * DIM + col],
                         W1[(2 * DIM + K + 1) * DIM + col]);
}

// ---------------- Kernel 2: pairwise head via WMMA (transposed) ----------------
// Each block: 64 pairs. hidden^T(128 x 64) = Wcd^T(128x256) @ X^T(256x64),
// tiled as 8 col-tiles (one per wave) x 4 pair-tiles x 8 k-steps of
// v_wmma_f32_16x16x32_bf16. C layout puts 8 columns of ONE pair in each lane,
// so the relu*W2 dot product reduces inside the lane.
__global__ __launch_bounds__(256)
void pair_head(const float* __restrict__ H,
               const unsigned char* __restrict__ legal_mask,
               const float* __restrict__ b1,
               const float* __restrict__ W2,
               const float* __restrict__ b2,
               const float* __restrict__ A,
               const float* __restrict__ Bm,
               const uint32_t* __restrict__ Wfrag,
               float* __restrict__ out) {
  __shared__ uint32_t Xs[PPB * XS_STRIDE];   // 64 pairs x 256 bf16 (dword-packed, padded)
  __shared__ int   pairI[PPB], pairJ[PPB], pmask[PPB];
  __shared__ float logit[PPB];

  const int tid  = threadIdx.x;
  const int l    = tid & 31;
  const int wv   = tid >> 5;            // wave id 0..7 -> 16-column tile of hidden
  const int half = l >> 4;              // dword K-half / C-row-half selector
  const long p0  = (long)blockIdx.x * PPB;

  // ---- decode 64 pair indices (closed form + fixup) ----
  if (tid < PPB) {
    const long p = p0 + tid;
    const double tn = (double)(2 * N_NODES - 1);
    int i = (int)((tn - sqrt(tn * tn - 8.0 * (double)p)) * 0.5);
    if (i < 0) i = 0;
    auto fcum = [](long ii) -> long {
      return ii * (N_NODES - 1) - (ii * (ii - 1)) / 2;
    };
    while (fcum(i + 1) <= p) ++i;
    while (fcum(i) > p) --i;
    const int j = (int)(i + 1 + (p - fcum(i)));
    pairI[tid] = i;
    pairJ[tid] = j;
    pmask[tid] = legal_mask[(long)i * N_NODES + j];
    logit[tid] = 0.f;
  }
  __syncthreads();

  // ---- build X tile: row p = [ |hi-hj| (k 0..127) , hi*hj (k 128..255) ] bf16 ----
  // 4 threads per pair, 32 features each; uint4 (b128) LDS stores.
  {
    const int p = tid >> 2;             // pair 0..63
    const int c = tid & 3;              // feature chunk 0..3 (32 floats)
    const float4* hi4 = (const float4*)(H + pairI[p] * DIM + c * 32);
    const float4* hj4 = (const float4*)(H + pairJ[p] * DIM + c * 32);
    uint4* rowAbs = (uint4*)&Xs[p * XS_STRIDE + c * 16];
    uint4* rowPrd = (uint4*)&Xs[p * XS_STRIDE + 64 + c * 16];
#pragma unroll
    for (int e = 0; e < 4; ++e) {
      float4 x0 = hi4[e * 2], x1 = hi4[e * 2 + 1];
      float4 y0 = hj4[e * 2], y1 = hj4[e * 2 + 1];
      uint4 da, pr;
      da.x = pack_bf16x2(fabsf(x0.x - y0.x), fabsf(x0.y - y0.y));
      da.y = pack_bf16x2(fabsf(x0.z - y0.z), fabsf(x0.w - y0.w));
      da.z = pack_bf16x2(fabsf(x1.x - y1.x), fabsf(x1.y - y1.y));
      da.w = pack_bf16x2(fabsf(x1.z - y1.z), fabsf(x1.w - y1.w));
      pr.x = pack_bf16x2(x0.x * y0.x, x0.y * y0.y);
      pr.y = pack_bf16x2(x0.z * y0.z, x0.w * y0.w);
      pr.z = pack_bf16x2(x1.x * y1.x, x1.y * y1.y);
      pr.w = pack_bf16x2(x1.z * y1.z, x1.w * y1.w);
      rowAbs[e] = da;
      rowPrd[e] = pr;
    }
  }

  // ---- preload this wave's 8 Wcd^T fragments (A operand; pre-packed, L2 resident) ----
  v16bf Wf[8];
  {
    const uint4* g = (const uint4*)Wfrag;
#pragma unroll
    for (int kk = 0; kk < 8; ++kk) {
      const int base = ((wv * 8 + kk) * 32 + l) * 2;   // uint4 units (8 dwords/lane)
      union { uint4 q[2]; v16bf v; } u;
      u.q[0] = g[base];
      u.q[1] = g[base + 1];
      Wf[kk] = u.v;
    }
  }
  __syncthreads();

  // ---- 32 x v_wmma_f32_16x16x32_bf16, 4 independent accumulator chains ----
  const int n16 = l & 15;               // B-operand column = pair slot within m-tile
  v8f acc[4];
#pragma unroll
  for (int mt = 0; mt < 4; ++mt) acc[mt] = (v8f){0.f,0.f,0.f,0.f,0.f,0.f,0.f,0.f};
#pragma unroll
  for (int kk = 0; kk < 8; ++kk) {
#pragma unroll
    for (int mt = 0; mt < 4; ++mt) {
      union { uint4 q[2]; v16bf v; } u;
      const uint4* s =
          (const uint4*)&Xs[(mt * 16 + n16) * XS_STRIDE + kk * 16 + half * 8];
      u.q[0] = s[0];
      u.q[1] = s[1];
      acc[mt] = __builtin_amdgcn_wmma_f32_16x16x32_bf16(
          false, Wf[kk], false, u.v, (short)0, acc[mt], false, false);
    }
  }

  // ---- epilogue: + A[i]+B[j]+b1, relu, *W2, in-lane reduce over 8 cols ----
  const int cbase = wv * 16 + half * 8;   // 8 consecutive hidden columns per lane
  float b1v[8], w2v[8];
  {
    float4 t0 = *(const float4*)(b1 + cbase);
    float4 t1 = *(const float4*)(b1 + cbase + 4);
    float4 t2 = *(const float4*)(W2 + cbase);
    float4 t3 = *(const float4*)(W2 + cbase + 4);
    b1v[0]=t0.x; b1v[1]=t0.y; b1v[2]=t0.z; b1v[3]=t0.w;
    b1v[4]=t1.x; b1v[5]=t1.y; b1v[6]=t1.z; b1v[7]=t1.w;
    w2v[0]=t2.x; w2v[1]=t2.y; w2v[2]=t2.z; w2v[3]=t2.w;
    w2v[4]=t3.x; w2v[5]=t3.y; w2v[6]=t3.z; w2v[7]=t3.w;
  }
#pragma unroll
  for (int mt = 0; mt < 4; ++mt) {
    const int m = mt * 16 + n16;        // pair slot 0..63
    const int i = pairI[m];
    const int j = pairJ[m];
    const float4* Ai = (const float4*)(A  + i * DIM + cbase);
    const float4* Bj = (const float4*)(Bm + j * DIM + cbase);
    float4 a0 = Ai[0], a1 = Ai[1];
    float4 q0 = Bj[0], q1 = Bj[1];
    float av[8] = {a0.x,a0.y,a0.z,a0.w,a1.x,a1.y,a1.z,a1.w};
    float qv[8] = {q0.x,q0.y,q0.z,q0.w,q1.x,q1.y,q1.z,q1.w};
    float part = 0.f;
#pragma unroll
    for (int r = 0; r < 8; ++r) {
      float h = acc[mt][r] + av[r] + qv[r] + b1v[r];
      h = fmaxf(h, 0.f);
      part = fmaf(h, w2v[r], part);
    }
    part += __shfl_xor(part, 16, 32);   // combine the two column-halves
    if (l < 16) atomicAdd(&logit[m], part);   // conflict-free ds_add_f32
  }
  __syncthreads();

  if (tid < PPB) {
    const float v = logit[tid] + b2[0];
    out[p0 + tid] = pmask[tid] ? v : 0.0f;
  }
}

// ---------------- launcher ----------------
extern "C" void kernel_launch(void* const* d_in, const int* in_sizes, int n_in,
                              void* d_out, int out_size, void* d_ws, size_t ws_size,
                              hipStream_t stream) {
  const float*         H    = (const float*)d_in[0];
  const unsigned char* mask = (const unsigned char*)d_in[1];
  const float*         W1   = (const float*)d_in[2];
  const float*         b1   = (const float*)d_in[3];
  const float*         W2   = (const float*)d_in[4];
  const float*         b2   = (const float*)d_in[5];
  float*               out  = (float*)d_out;

  // workspace layout: A (N*D f32) | B (N*D f32) | Wfrag (16384 dwords)
  float*    A     = (float*)d_ws;
  float*    B     = A + N_NODES * DIM;
  uint32_t* Wfrag = (uint32_t*)(B + N_NODES * DIM);

  precompute_AB<<<N_NODES, DIM, 0, stream>>>(H, W1, A, B);
  pack_wfrag<<<64, 256, 0, stream>>>(W1, Wfrag);
  pair_head<<<PAIRS / PPB, 256, 0, stream>>>(H, mask, b1, W2, b2, A, B, Wfrag, out);
}
